// Qwen35GatedDeltaNet_63359357551115
// MI455X (gfx1250) — matrix operand order repaired and checked by reading, not compiled, not run on hardware
//
#include <hip/hip_runtime.h>
#include <hip/hip_bf16.h>
#include <math.h>

// ---- problem constants (Qwen3.5 GatedDeltaNet block) ----
#define BB      2
#define SS      1024
#define HIDDEN_ 2048
#define HV_     32
#define HK_     16
#define DK_     128
#define DV_     128
#define KEY_DIM_  2048
#define VAL_DIM_  4096
#define CONV_DIM_ 8192
#define BS_     (BB*SS)          // 2048 rows

typedef __attribute__((ext_vector_type(16))) __bf16 v16bf;
typedef __attribute__((ext_vector_type(8)))  float  v8f;

__device__ __forceinline__ unsigned short f2bf(float x) {
  unsigned u = __float_as_uint(x);
  u = (u + 0x7fffu + ((u >> 16) & 1u)) >> 16;   // round-to-nearest-even
  return (unsigned short)u;
}

__device__ __forceinline__ void wait_async0() {
#if __has_builtin(__builtin_amdgcn_s_wait_asynccnt)
  __builtin_amdgcn_s_wait_asynccnt(0);
#else
  asm volatile("s_wait_asynccnt 0" ::: "memory");
#endif
}

// ---------------- fp32 -> bf16 conversion ----------------
__global__ void cvt_bf16_kernel(const float* __restrict__ in,
                                unsigned short* __restrict__ out, int n) {
  int i = blockIdx.x * blockDim.x + threadIdx.x;
  if (i < n) out[i] = f2bf(in[i]);
}

// ---------------- bf16 WMMA GEMM: C[M,N] = A[M,K] @ B[K,N] ----------------
// 256 threads = 8 waves; 128x128 C tile; BK=32; fp32 accumulate.
// A tile: async global->LDS copy (ASYNCcnt) issued for tile k+1 while WMMAs
// consume tile k. B tile: coalesced b32 loads, transpose deferred to b16 LDS
// stores. Double-buffered LDS, b128 fragment loads, prefetch for tile k+2.
#define LSTR 20   // padded LDS row stride in dwords (rows 16B aligned, bank-clean)

__launch_bounds__(256)
__global__ void gemm_bf16_kernel(const unsigned short* __restrict__ A,
                                 const unsigned short* __restrict__ Bm,
                                 float* __restrict__ C,
                                 int M, int N, int K) {
  __shared__ __align__(16) unsigned int Asm_[2][128 * LSTR];
  __shared__ __align__(16) unsigned int Bsm_[2][128 * LSTR];
  const int tid   = threadIdx.x;
  const int lane  = tid & 31;
  const int wave  = tid >> 5;
  const int waveM = wave & 1;        // 0..1 -> 64-row half
  const int waveN = wave >> 1;       // 0..3 -> 32-col slice
  const int half  = lane >> 4;       // lane half
  const int l16   = lane & 15;
  const int m0 = blockIdx.y * 128;
  const int n0 = blockIdx.x * 128;
  const unsigned int* Aui = (const unsigned int*)A;
  const unsigned int* Bui = (const unsigned int*)Bm;

  v8f acc[4][2];
  #pragma unroll
  for (int mi = 0; mi < 4; mi++)
    #pragma unroll
    for (int ni = 0; ni < 2; ni++)
      #pragma unroll
      for (int r = 0; r < 8; r++) acc[mi][ni][r] = 0.f;

  // A tile: 128 rows x 32 halves = 2048 dwords, async copy global -> LDS
  auto asyncLoadA = [&](int buf, int k0) {
    #pragma unroll
    for (int i = 0; i < 8; i++) {
      int idx = tid + i * 256;
      int row = idx >> 4, kd = idx & 15;
      unsigned lds  = (unsigned)(unsigned long long)&Asm_[buf][row * LSTR + kd];
      unsigned voff = (unsigned)(((m0 + row) * K + k0) * 2 + kd * 4);  // bytes
      asm volatile("global_load_async_to_lds_b32 %0, %1, %2"
                   :: "v"(lds), "v"(voff), "s"(A) : "memory");
    }
  };
  // B tile: 32 rows x 128 cols; coalesced dword loads (2 columns per dword)
  unsigned rb[8];
  auto loadB = [&](int k0) {
    #pragma unroll
    for (int i = 0; i < 8; i++) {
      int idx = tid + i * 256;              // 0..2047
      int kk = idx >> 6, np = idx & 63;     // row k0+kk, dword-column np
      rb[i] = Bui[(((k0 + kk) * N + n0) >> 1) + np];
    }
  };
  // transpose on the LDS-store side: two b16 stores per dword
  auto storeB = [&](int buf) {
    unsigned short* Bh = (unsigned short*)&Bsm_[buf][0];
    #pragma unroll
    for (int i = 0; i < 8; i++) {
      int idx = tid + i * 256;
      int kk = idx >> 6, np = idx & 63;
      Bh[(2 * np) * (2 * LSTR) + kk]     = (unsigned short)rb[i];
      Bh[(2 * np + 1) * (2 * LSTR) + kk] = (unsigned short)(rb[i] >> 16);
    }
  };
  // 8 WMMAs on one LDS buffer
  auto compute = [&](int buf) {
    union { uint4 q[2]; unsigned u[8]; v16bf v; } bfrag[2];
    #pragma unroll
    for (int ni = 0; ni < 2; ni++) {
      int n = waveN * 32 + ni * 16 + l16;
      bfrag[ni].q[0] = *(const uint4*)&Bsm_[buf][n * LSTR + half * 4];
      bfrag[ni].q[1] = *(const uint4*)&Bsm_[buf][n * LSTR + 8 + half * 4];
    }
    #pragma unroll
    for (int mi = 0; mi < 4; mi++) {
      union { uint4 q[2]; unsigned u[8]; v16bf v; } afrag;
      int row = waveM * 64 + mi * 16 + l16;
      afrag.q[0] = *(const uint4*)&Asm_[buf][row * LSTR + half * 4];
      afrag.q[1] = *(const uint4*)&Asm_[buf][row * LSTR + 8 + half * 4];
      #pragma unroll
      for (int ni = 0; ni < 2; ni++) {
        acc[mi][ni] = __builtin_amdgcn_wmma_f32_16x16x32_bf16(
            false, afrag.v, false, bfrag[ni].v, (short)0, acc[mi][ni],
            false, false);
      }
    }
  };

  const int nk = K >> 5;
  asyncLoadA(0, 0);
  loadB(0);
  storeB(0);
  wait_async0();
  __syncthreads();
  for (int kt = 0; kt < nk; kt++) {
    int cur = kt & 1;
    if (kt + 1 < nk) {
      asyncLoadA(1 - cur, (kt + 1) << 5);
      loadB((kt + 1) << 5);
      if (kt + 2 < nk) {  // global_prefetch_b8 for tile k+2
        int k2 = (kt + 2) << 5;
        __builtin_prefetch(&Aui[(((m0 + (tid >> 4)) * K + k2) >> 1) + (tid & 15)], 0, 0);
        __builtin_prefetch(&Bui[(((k2 + (tid >> 6)) * N + n0) >> 1) + (tid & 63)], 0, 0);
      }
    }
    compute(cur);
    if (kt + 1 < nk) storeB(1 - cur);
    wait_async0();          // async A-tile for next iter is in LDS
    __syncthreads();
  }

  // store C per documented layout: VGPR r -> M = half*8 + r, N = l16
  #pragma unroll
  for (int mi = 0; mi < 4; mi++)
    #pragma unroll
    for (int ni = 0; ni < 2; ni++)
      #pragma unroll
      for (int r = 0; r < 8; r++) {
        long gm = m0 + waveM * 64 + mi * 16 + half * 8 + r;
        long gn = n0 + waveN * 32 + ni * 16 + l16;
        C[gm * (long)N + gn] = acc[mi][ni][r];
      }
}

// ---------------- skinny b/a projection + beta/g activations ----------------
__global__ void ba_kernel(const float* __restrict__ H, const float* __restrict__ wb,
                          const float* __restrict__ wa, const float* __restrict__ dtb,
                          const float* __restrict__ A_log,
                          float* __restrict__ beta, float* __restrict__ g) {
  int bs = blockIdx.x;          // 0..BS-1
  int h  = threadIdx.x;         // 0..31
  const float* hrow = H + (long)bs * HIDDEN_;
  float accb = 0.f, acca = 0.f;
  for (int k = 0; k < HIDDEN_; k++) {
    float x = hrow[k];                 // uniform address -> broadcast
    accb += x * wb[k * HV_ + h];
    acca += x * wa[k * HV_ + h];
  }
  beta[bs * HV_ + h] = 1.f / (1.f + __expf(-accb));
  float sx = acca + dtb[h];
  float sp = sx > 0.f ? sx + log1pf(__expf(-sx)) : log1pf(__expf(sx));
  g[bs * HV_ + h] = -__expf(A_log[h]) * sp;
}

// ---------------- causal depthwise conv1d (K=4) + SiLU ----------------
__global__ void conv_silu_kernel(const float* __restrict__ x, const float* __restrict__ w,
                                 float* __restrict__ y) {
  long gid = (long)blockIdx.x * blockDim.x + threadIdx.x;
  int c = (int)(gid % CONV_DIM_);
  long rest = gid / CONV_DIM_;
  int sb = (int)(rest % (SS / 64));
  int b  = (int)(rest / (SS / 64));
  float w0 = w[c * 4 + 0], w1 = w[c * 4 + 1], w2 = w[c * 4 + 2], w3 = w[c * 4 + 3];
  int s0 = sb * 64;
  const float* xb = x + (long)b * SS * CONV_DIM_ + c;
  float xm3 = (s0 >= 3) ? xb[(long)(s0 - 3) * CONV_DIM_] : 0.f;
  float xm2 = (s0 >= 2) ? xb[(long)(s0 - 2) * CONV_DIM_] : 0.f;
  float xm1 = (s0 >= 1) ? xb[(long)(s0 - 1) * CONV_DIM_] : 0.f;
  for (int s = s0; s < s0 + 64; s++) {
    float xc = xb[(long)s * CONV_DIM_];
    float v = w0 * xm3 + w1 * xm2 + w2 * xm1 + w3 * xc;
    v = v / (1.f + __expf(-v));        // SiLU
    y[((long)b * SS + s) * CONV_DIM_ + c] = v;
    xm3 = xm2; xm2 = xm1; xm1 = xc;
  }
}

// ---------------- q/k L2 norm (+ q * DK^-0.5), one wave per row ----------------
__global__ void qknorm_kernel(const float* __restrict__ convout,
                              float* __restrict__ qn, float* __restrict__ kn) {
  int tid = threadIdx.x;
  int wv = tid >> 5, lane = tid & 31;
  long row = (long)blockIdx.x * 8 + wv;           // 0 .. 2*BS*HK-1
  int isK = row >= (long)BS_ * HK_;
  long r  = isK ? row - (long)BS_ * HK_ : row;
  long bs = r / HK_;
  int  h  = (int)(r % HK_);
  const float* src = convout + bs * CONV_DIM_ + (isK ? KEY_DIM_ : 0) + h * DK_ + lane * 4;
  float4 v = *(const float4*)src;
  float ss = v.x * v.x + v.y * v.y + v.z * v.z + v.w * v.w;
  #pragma unroll
  for (int off = 16; off >= 1; off >>= 1) ss += __shfl_xor(ss, off, 32);
  float scale = rsqrtf(ss + 1e-6f) * (isK ? 1.f : 0.0883883476483184f); // q: *DK^-1/2
  float* dst = (isK ? kn : qn) + (bs * HK_ + h) * DK_ + lane * 4;
  float4 o; o.x = v.x * scale; o.y = v.y * scale; o.z = v.z * scale; o.w = v.w * scale;
  *(float4*)dst = o;
}

// ---------------- gated delta-rule recurrence ----------------
// One 128-thread block per (b, hv). State S[DK,DV] fp32: lane owns column dv=tid,
// 128 rows held in VGPRs. q/k chunks staged in LDS (broadcast reads).
__launch_bounds__(128)
__global__ void recur_kernel(const float* __restrict__ qn, const float* __restrict__ kn,
                             const float* __restrict__ convout,
                             const float* __restrict__ g, const float* __restrict__ beta,
                             float* __restrict__ o) {
  __shared__ float qs[32][DK_];
  __shared__ float ks[32][DK_];
  __shared__ float gsh[32];
  __shared__ float bsh[32];
  int b  = blockIdx.x / HV_;
  int hv = blockIdx.x % HV_;
  int hk = hv >> 1;                    // GQA rep = 2
  int tid = threadIdx.x;

  float St[DK_];
  #pragma unroll
  for (int i = 0; i < DK_; i++) St[i] = 0.f;

  for (int c0 = 0; c0 < SS; c0 += 32) {
    __syncthreads();
    for (int t = 0; t < 32; t++) {
      long bsrow = (long)b * SS + c0 + t;
      qs[t][tid] = qn[(bsrow * HK_ + hk) * DK_ + tid];
      ks[t][tid] = kn[(bsrow * HK_ + hk) * DK_ + tid];
    }
    if (tid < 32) {
      long bsrow = (long)b * SS + c0 + tid;
      gsh[tid] = g[bsrow * HV_ + hv];
      bsh[tid] = beta[bsrow * HV_ + hv];
    }
    if (c0 + 32 < SS) {   // prefetch next chunk (global_prefetch_b8)
      long nrow = (long)b * SS + c0 + 32;
      __builtin_prefetch(&qn[(nrow * HK_ + hk) * DK_ + tid], 0, 0);
      __builtin_prefetch(&kn[(nrow * HK_ + hk) * DK_ + tid], 0, 0);
    }
    __syncthreads();

    for (int t = 0; t < 32; t++) {
      float e = __expf(gsh[t]);
      float kv = 0.f;
      #pragma unroll
      for (int kk = 0; kk < DK_; kk++) kv += ks[t][kk] * St[kk];
      kv *= e;                               // readout of decayed state
      long bsrow = (long)b * SS + c0 + t;
      float vv = convout[bsrow * CONV_DIM_ + 2 * KEY_DIM_ + hv * DV_ + tid];
      float delta = (vv - kv) * bsh[t];
      float ot = 0.f;
      #pragma unroll
      for (int kk = 0; kk < DK_; kk++) {
        St[kk] = St[kk] * e + ks[t][kk] * delta;   // decay + rank-1 write
        ot += qs[t][kk] * St[kk];                  // query readout
      }
      o[(bsrow * HV_ + hv) * DV_ + tid] = ot;
    }
  }
}

// ---------------- gated RMSNorm -> bf16 for final GEMM ----------------
__global__ void gatenorm_kernel(const float* __restrict__ o, const float* __restrict__ z,
                                const float* __restrict__ nw,
                                unsigned short* __restrict__ onb) {
  int tid = threadIdx.x;
  int wv = tid >> 5, lane = tid & 31;
  long row = (long)blockIdx.x * 8 + wv;      // 0 .. BS*HV-1
  long base = row * DV_ + lane * 4;
  float gz[4];
  float ss = 0.f;
  #pragma unroll
  for (int i = 0; i < 4; i++) {
    float ov = o[base + i];
    float zv = z[base + i];
    float s  = zv / (1.f + __expf(-zv));     // silu(z)
    gz[i] = ov * s;
    ss += gz[i] * gz[i];
  }
  #pragma unroll
  for (int off = 16; off >= 1; off >>= 1) ss += __shfl_xor(ss, off, 32);
  float inv = rsqrtf(ss * (1.f / DV_) + 1e-6f);
  #pragma unroll
  for (int i = 0; i < 4; i++)
    onb[base + i] = f2bf(gz[i] * inv * nw[lane * 4 + i]);
}

// ---------------- launch ----------------
extern "C" void kernel_launch(void* const* d_in, const int* in_sizes, int n_in,
                              void* d_out, int out_size, void* d_ws, size_t ws_size,
                              hipStream_t stream) {
  const float* H      = (const float*)d_in[0];
  const float* conv_w = (const float*)d_in[1];
  const float* w_qkv  = (const float*)d_in[2];
  const float* w_z    = (const float*)d_in[3];
  const float* w_b    = (const float*)d_in[4];
  const float* w_a    = (const float*)d_in[5];
  const float* w_out  = (const float*)d_in[6];
  const float* dt_b   = (const float*)d_in[7];
  const float* A_log  = (const float*)d_in[8];
  const float* norm_w = (const float*)d_in[9];
  float* out = (float*)d_out;

  char* ws = (char*)d_ws;
  size_t off = 0;
  auto alloc = [&](size_t bytes) -> char* {
    char* p = ws + off; off += (bytes + 255) & ~(size_t)255; return p;
  };
  float* mixed   = (float*)alloc((size_t)BS_ * CONV_DIM_ * 4);   // 64 MB
  float* convout = (float*)alloc((size_t)BS_ * CONV_DIM_ * 4);   // 64 MB
  float* z       = (float*)alloc((size_t)BS_ * VAL_DIM_ * 4);    // 32 MB
  float* o       = (float*)alloc((size_t)BS_ * VAL_DIM_ * 4);    // 32 MB
  float* qn      = (float*)alloc((size_t)BS_ * KEY_DIM_ * 4);    // 16 MB
  float* kn      = (float*)alloc((size_t)BS_ * KEY_DIM_ * 4);    // 16 MB
  unsigned short* Hb    = (unsigned short*)alloc((size_t)BS_ * HIDDEN_ * 2);
  unsigned short* Wqkvb = (unsigned short*)alloc((size_t)HIDDEN_ * CONV_DIM_ * 2);
  unsigned short* Wzb   = (unsigned short*)alloc((size_t)HIDDEN_ * VAL_DIM_ * 2);
  unsigned short* Woutb = (unsigned short*)alloc((size_t)VAL_DIM_ * HIDDEN_ * 2);
  unsigned short* onb   = (unsigned short*)alloc((size_t)BS_ * VAL_DIM_ * 2);
  float* beta = (float*)alloc((size_t)BS_ * HV_ * 4);
  float* g    = (float*)alloc((size_t)BS_ * HV_ * 4);

  auto cvt = [&](const float* src, unsigned short* dst, long n) {
    cvt_bf16_kernel<<<(int)((n + 255) / 256), 256, 0, stream>>>(src, dst, (int)n);
  };
  cvt(H,     Hb,    (long)BS_ * HIDDEN_);
  cvt(w_qkv, Wqkvb, (long)HIDDEN_ * CONV_DIM_);
  cvt(w_z,   Wzb,   (long)HIDDEN_ * VAL_DIM_);
  cvt(w_out, Woutb, (long)VAL_DIM_ * HIDDEN_);

  // big GEMMs on the matrix pipe
  gemm_bf16_kernel<<<dim3(CONV_DIM_ / 128, BS_ / 128), 256, 0, stream>>>(
      Hb, Wqkvb, mixed, BS_, CONV_DIM_, HIDDEN_);
  gemm_bf16_kernel<<<dim3(VAL_DIM_ / 128, BS_ / 128), 256, 0, stream>>>(
      Hb, Wzb, z, BS_, VAL_DIM_, HIDDEN_);

  ba_kernel<<<BS_, 32, 0, stream>>>(H, w_b, w_a, dt_b, A_log, beta, g);

  conv_silu_kernel<<<(BB * (SS / 64) * CONV_DIM_) / 256, 256, 0, stream>>>(
      mixed, conv_w, convout);

  qknorm_kernel<<<(2 * BS_ * HK_) / 8, 256, 0, stream>>>(convout, qn, kn);

  recur_kernel<<<BB * HV_, 128, 0, stream>>>(qn, kn, convout, g, beta, o);

  gatenorm_kernel<<<(BS_ * HV_) / 8, 256, 0, stream>>>(o, z, norm_w, onb);

  gemm_bf16_kernel<<<dim3(HIDDEN_ / 128, BS_ / 128), 256, 0, stream>>>(
      onb, Woutb, out, BS_, HIDDEN_, VAL_DIM_);
}
